// SentenceEncoder_30751965839500
// MI455X (gfx1250) — matrix operand order repaired and checked by reading, not compiled, not run on hardware
//
#include <hip/hip_runtime.h>
#include <hip/hip_bf16.h>
#include <cstdint>

// ---------------- problem constants ----------------
constexpr int BB   = 8;
constexpr int SS   = 1024;
constexpr int DD   = 768;
constexpr int LL   = 12;
constexpr int HH   = 12;
constexpr int FFD  = 3072;
constexpr int HDIM = 64;
constexpr int MR   = BB * SS;      // 8192 rows of activations

typedef __attribute__((ext_vector_type(16))) __bf16 v16bf;
typedef __attribute__((ext_vector_type(8)))  float  v8f;

union Frag16 { v16bf v; unsigned u[8]; };

__device__ inline v8f wmma_bf16(v16bf a, v16bf b, v8f c) {
  // (neg_a, A, neg_b, B, c_mod, C, reuse_a, reuse_b)
  return __builtin_amdgcn_wmma_f32_16x16x32_bf16(false, a, false, b, (short)0, c, false, false);
}

// A-matrix (16x32 bf16) K-pair base for vgpr index vv, lane-group g (contiguous k = 16g + 2vv, matching B)
__device__ inline int a_kpair(int vv, int g) {
  return 2 * vv + 16 * g;
}

// ---- CDNA5 async global->LDS copy (ASYNCcnt path, ISA 10.x / 15.18.3) ----
// LDS destination offset = low 32 bits of the generic pointer to __shared__
// (addrspace(3) offsets occupy the low dword; hardware adds LDS_BASE).
__device__ inline void async_copy_b128(void* lds_dst, const void* gsrc) {
  uint32_t lo = (uint32_t)(uintptr_t)lds_dst;
  uint64_t ga = (uint64_t)(uintptr_t)gsrc;
  asm volatile("global_load_async_to_lds_b128 %0, %1, off"
               :: "v"(lo), "v"(ga) : "memory");
}
__device__ inline void wait_async0() {
  asm volatile("s_wait_asynccnt 0" ::: "memory");
}

__device__ inline float block_sum256(float x, volatile float* sc) {
#pragma unroll
  for (int off = 16; off > 0; off >>= 1) x += __shfl_xor(x, off);
  int wave = threadIdx.x >> 5;
  if ((threadIdx.x & 31) == 0) sc[wave] = x;
  __syncthreads();
  float r = 0.f;
#pragma unroll
  for (int i = 0; i < 8; ++i) r += sc[i];
  __syncthreads();
  return r;
}

// ---------------- weight conversion ----------------
__global__ __launch_bounds__(256) void f32_to_bf16_k(const float* __restrict__ x,
                                                     __bf16* __restrict__ y, size_t n) {
  size_t i = (size_t)blockIdx.x * 256u + threadIdx.x;
  size_t stride = (size_t)gridDim.x * 256u;
  for (; i < n; i += stride) y[i] = (__bf16)x[i];
}

// ---------------- embeddings + LayerNorm ----------------
__global__ __launch_bounds__(256) void embed_ln_k(
    const int* __restrict__ ids, const float* __restrict__ wemb,
    const float* __restrict__ pemb, const float* __restrict__ temb,
    const float* __restrict__ ls, const float* __restrict__ lb,
    float* __restrict__ hF, __bf16* __restrict__ hB) {
  __shared__ float sc0[8];
  __shared__ float sc1[8];
  int row = blockIdx.x;               // 0..MR-1
  int s  = row & (SS - 1);
  int tok = ids[row];
  float vals[3]; float sum = 0.f, sq = 0.f;
#pragma unroll
  for (int i = 0; i < 3; ++i) {
    int d = threadIdx.x + i * 256;
    float x = wemb[(size_t)tok * DD + d] + pemb[(size_t)s * DD + d] + temb[d];
    vals[i] = x; sum += x; sq += x * x;
  }
  sum = block_sum256(sum, sc0);
  sq  = block_sum256(sq,  sc1);
  float mean = sum * (1.f / DD);
  float var  = sq * (1.f / DD) - mean * mean;
  float inv  = rsqrtf(var + 1e-12f);
#pragma unroll
  for (int i = 0; i < 3; ++i) {
    int d = threadIdx.x + i * 256;
    float y = (vals[i] - mean) * inv * ls[d] + lb[d];
    hF[(size_t)row * DD + d] = y;
    hB[(size_t)row * DD + d] = (__bf16)y;
  }
}

// ---------------- bf16 GEMM: C[M,N] = A[M,K] @ W[K,N] + bias ----------------
// block: 128 threads (4 waves), tile 64x64, k-step 32.
// A tile staged via async global->LDS (no transpose needed); W tile staged
// through VGPRs with an LDS-side transpose so WMMA B-fragment K-pairs are
// contiguous (2x ds_load_b128 per fragment).
template <int ACT, bool OUTF, bool OUTB>
__global__ __launch_bounds__(128) void gemm_bf16_k(
    const __bf16* __restrict__ A, const __bf16* __restrict__ W,
    const float* __restrict__ bias, float* __restrict__ outF,
    __bf16* __restrict__ outB, int M, int N, int K) {
  __shared__ __bf16 At[64][32];     // [row][k]
  __shared__ __bf16 Bt[64][32];     // transposed: [n][k]
  int t = threadIdx.x, wave = t >> 5, lane = t & 31;
  int m0 = blockIdx.x * 64, n0 = blockIdx.y * 64;
  int g = lane >> 4, ln = lane & 15;

  v8f acc[4] = {v8f{}, v8f{}, v8f{}, v8f{}};

  for (int k0 = 0; k0 < K; k0 += 32) {
    { // A tile: thread -> row t>>1, 16 halves at (t&1)*16 (async to LDS)
      int r = t >> 1, c = (t & 1) * 16;
      const __bf16* src = A + (size_t)(m0 + r) * K + k0 + c;
      async_copy_b128(&At[r][c],     src);
      async_copy_b128(&At[r][c + 8], src + 8);
      __builtin_prefetch((const void*)(src + 32), 0, 1);   // next k-tile
    }
    { // B tile (transposed store): thread -> W row kk=t>>2, 16 cols at (t&3)*16
      int kk = t >> 2, c = (t & 3) * 16;
      const __bf16* src = W + (size_t)(k0 + kk) * N + n0 + c;
      __bf16 tmp[16];
      *(uint4*)&tmp[0] = *(const uint4*)src;
      *(uint4*)&tmp[8] = *(const uint4*)(src + 8);
#pragma unroll
      for (int i = 0; i < 16; ++i) Bt[c + i][kk] = tmp[i];
      __builtin_prefetch((const void*)(src + (size_t)32 * N), 0, 1);
    }
    wait_async0();
    __syncthreads();

    Frag16 af;
    {
      int mrow = ln + wave * 16;
#pragma unroll
      for (int vv = 0; vv < 8; ++vv)
        af.u[vv] = *(const unsigned*)&At[mrow][a_kpair(vv, g)];
    }
#pragma unroll
    for (int ns = 0; ns < 4; ++ns) {
      Frag16 bfr;
      int n = ns * 16 + ln;
#pragma unroll
      for (int vv = 0; vv < 8; ++vv)
        bfr.u[vv] = *(const unsigned*)&Bt[n][2 * vv + 16 * g];
      acc[ns] = wmma_bf16(af.v, bfr.v, acc[ns]);
    }
    __syncthreads();
  }

  // epilogue: bias (+GELU), compile-time-selected stores
#pragma unroll
  for (int ns = 0; ns < 4; ++ns) {
    int col = n0 + ns * 16 + ln;
    float bv = bias[col];
#pragma unroll
    for (int r = 0; r < 8; ++r) {
      int row = m0 + wave * 16 + r + 8 * g;
      float x = acc[ns][r] + bv;
      if (ACT == 1) x = 0.5f * x * (1.0f + erff(x * 0.70710678118f));
      if (OUTF) outF[(size_t)row * N + col] = x;
      if (OUTB) outB[(size_t)row * N + col] = (__bf16)x;
    }
  }
}

// ---------------- flash attention (per (b,h), 128 q-rows / block) ----------------
__global__ __launch_bounds__(256) void attention_k(
    const __bf16* __restrict__ qp, const __bf16* __restrict__ kp,
    const __bf16* __restrict__ vp, const int* __restrict__ amask,
    __bf16* __restrict__ ctx) {
  __shared__ __bf16 kt[32][HDIM];        // [key][hd]  (async-loaded, row-major)
  __shared__ __bf16 vt[HDIM][32];        // [hd][key]  (transposed in LDS)
  __shared__ __bf16 pbuf[8][16][32];     // per-wave P repack buffer
  int b = blockIdx.z, h = blockIdx.y;
  int t = threadIdx.x, wave = t >> 5, lane = t & 31;
  int g = lane >> 4, ln = lane & 15;
  int q0 = blockIdx.x * 128 + wave * 16;

  // Q fragments (A layout, hd 0..31 and 32..63), read directly from global
  Frag16 qf0, qf1;
  {
    const __bf16* qrow = qp + ((size_t)b * SS + q0 + ln) * DD + h * HDIM;
#pragma unroll
    for (int vv = 0; vv < 8; ++vv) {
      int kb = a_kpair(vv, g);
      qf0.u[vv] = *(const unsigned*)(qrow + kb);
      qf1.u[vv] = *(const unsigned*)(qrow + 32 + kb);
    }
  }

  float rmax[8], rsum[8];
#pragma unroll
  for (int r = 0; r < 8; ++r) { rmax[r] = -1e30f; rsum[r] = 0.f; }
  v8f acc[4] = {v8f{}, v8f{}, v8f{}, v8f{}};
  const float scale = 0.125f;   // 1/sqrt(64)

  for (int j = 0; j < SS; j += 32) {
    { // cooperative K/V chunk load: K async to LDS, V transposed via VGPRs
      int key = t >> 3, c = (t & 7) * 8;
      size_t base = ((size_t)b * SS + j + key) * DD + h * HDIM + c;
      async_copy_b128(&kt[key][c], kp + base);
      __bf16 tmp[8];
      *(uint4*)tmp = *(const uint4*)(vp + base);
#pragma unroll
      for (int i = 0; i < 8; ++i) vt[c + i][key] = tmp[i];
      __builtin_prefetch((const void*)(kp + base + (size_t)32 * DD), 0, 1);
      __builtin_prefetch((const void*)(vp + base + (size_t)32 * DD), 0, 1);
    }
    wait_async0();
    __syncthreads();

    // scores for two 16-key subtiles: s = Q(16x64) @ K^T(64x16)
    v8f s[2];
#pragma unroll
    for (int sub = 0; sub < 2; ++sub) {
      Frag16 b0, b1;
      int key = sub * 16 + ln;
#pragma unroll
      for (int vv = 0; vv < 8; ++vv) {
        int kb = 2 * vv + 16 * g;
        b0.u[vv] = *(const unsigned*)&kt[key][kb];
        b1.u[vv] = *(const unsigned*)&kt[key][32 + kb];
      }
      v8f c0 = {};
      c0 = wmma_bf16(qf0.v, b0.v, c0);
      c0 = wmma_bf16(qf1.v, b1.v, c0);
      s[sub] = c0;
    }

    // scale + mask (pre-softmax mask == reference post-softmax mask + renorm)
#pragma unroll
    for (int sub = 0; sub < 2; ++sub) {
      int key = j + sub * 16 + ln;
      int mk = amask[b * SS + key];
#pragma unroll
      for (int r = 0; r < 8; ++r) {
        float x = s[sub][r] * scale;
        s[sub][r] = mk ? x : -1e30f;
      }
    }

    // online softmax: tile max per row (reduce over 16 lanes of each half-wave)
    float tmax[8];
#pragma unroll
    for (int r = 0; r < 8; ++r) tmax[r] = fmaxf(s[0][r], s[1][r]);
#pragma unroll
    for (int off = 1; off < 16; off <<= 1)
#pragma unroll
      for (int r = 0; r < 8; ++r) tmax[r] = fmaxf(tmax[r], __shfl_xor(tmax[r], off));

    float alpha[8];
#pragma unroll
    for (int r = 0; r < 8; ++r) {
      float mnew = fmaxf(rmax[r], tmax[r]);
      alpha[r] = __expf(rmax[r] - mnew);
      rmax[r] = mnew;
    }
    float psum[8];
#pragma unroll
    for (int r = 0; r < 8; ++r) psum[r] = 0.f;
#pragma unroll
    for (int sub = 0; sub < 2; ++sub)
#pragma unroll
      for (int r = 0; r < 8; ++r) {
        float p = __expf(s[sub][r] - rmax[r]);
        s[sub][r] = p; psum[r] += p;
      }
#pragma unroll
    for (int off = 1; off < 16; off <<= 1)
#pragma unroll
      for (int r = 0; r < 8; ++r) psum[r] += __shfl_xor(psum[r], off);
#pragma unroll
    for (int r = 0; r < 8; ++r) rsum[r] = rsum[r] * alpha[r] + psum[r];

    // rescale ctx accumulators (row index r maps identically in C frags)
#pragma unroll
    for (int ns = 0; ns < 4; ++ns)
#pragma unroll
      for (int r = 0; r < 8; ++r) acc[ns][r] *= alpha[r];

    // repack P from C layout -> A layout via per-wave LDS
#pragma unroll
    for (int sub = 0; sub < 2; ++sub)
#pragma unroll
      for (int r = 0; r < 8; ++r)
        pbuf[wave][r + 8 * g][sub * 16 + ln] = (__bf16)s[sub][r];
    Frag16 pf;
#pragma unroll
    for (int vv = 0; vv < 8; ++vv)
      pf.u[vv] = *(const unsigned*)&pbuf[wave][ln][a_kpair(vv, g)];

    // ctx += P(16x32) @ V(32x64)
#pragma unroll
    for (int ns = 0; ns < 4; ++ns) {
      Frag16 vf;
      int n = ns * 16 + ln;
#pragma unroll
      for (int vv = 0; vv < 8; ++vv)
        vf.u[vv] = *(const unsigned*)&vt[n][2 * vv + 16 * g];
      acc[ns] = wmma_bf16(pf.v, vf.v, acc[ns]);
    }
    __syncthreads();
  }

  // finalize: divide by row sums, store bf16 ctx in [B,S,D] (head-interleaved)
#pragma unroll
  for (int ns = 0; ns < 4; ++ns)
#pragma unroll
    for (int r = 0; r < 8; ++r) {
      int row = q0 + r + 8 * g;
      float y = acc[ns][r] / fmaxf(rsum[r], 1e-9f);
      ctx[((size_t)b * SS + row) * DD + h * HDIM + ns * 16 + ln] = (__bf16)y;
    }
}

// ---------------- residual + LayerNorm ----------------
__global__ __launch_bounds__(256) void residual_ln_k(
    const float* __restrict__ a, const float* __restrict__ resid,
    const float* __restrict__ ls, const float* __restrict__ lb,
    float* __restrict__ outF, __bf16* __restrict__ outB) {
  __shared__ float sc0[8];
  __shared__ float sc1[8];
  int row = blockIdx.x;
  float vals[3]; float sum = 0.f, sq = 0.f;
#pragma unroll
  for (int i = 0; i < 3; ++i) {
    int d = threadIdx.x + i * 256;
    float x = a[(size_t)row * DD + d] + resid[(size_t)row * DD + d];
    vals[i] = x; sum += x; sq += x * x;
  }
  sum = block_sum256(sum, sc0);
  sq  = block_sum256(sq,  sc1);
  float mean = sum * (1.f / DD);
  float var  = sq * (1.f / DD) - mean * mean;
  float inv  = rsqrtf(var + 1e-12f);
#pragma unroll
  for (int i = 0; i < 3; ++i) {
    int d = threadIdx.x + i * 256;
    float y = (vals[i] - mean) * inv * ls[d] + lb[d];
    outF[(size_t)row * DD + d] = y;
    outB[(size_t)row * DD + d] = (__bf16)y;
  }
}

// ---------------- masked mean pool + L2 normalize ----------------
__global__ __launch_bounds__(256) void pool_norm_k(
    const float* __restrict__ h, const int* __restrict__ amask,
    float* __restrict__ out) {
  __shared__ float sc[8];
  int b = blockIdx.x;
  float acc[3] = {0.f, 0.f, 0.f};
  float cnt = 0.f;
  for (int s = 0; s < SS; ++s) {
    float m = (float)amask[b * SS + s];
    cnt += m;
    const float* row = h + ((size_t)b * SS + s) * DD;
#pragma unroll
    for (int i = 0; i < 3; ++i) acc[i] += row[threadIdx.x + i * 256] * m;
  }
  float invc = 1.f / fmaxf(cnt, 1e-6f);
  float n2 = 0.f;
#pragma unroll
  for (int i = 0; i < 3; ++i) { acc[i] *= invc; n2 += acc[i] * acc[i]; }
  n2 = block_sum256(n2, sc);
  float inv = 1.f / fmaxf(sqrtf(n2), 1e-12f);
#pragma unroll
  for (int i = 0; i < 3; ++i)
    out[(size_t)b * DD + threadIdx.x + i * 256] = acc[i] * inv;
}

// ---------------- host orchestration ----------------
extern "C" void kernel_launch(void* const* d_in, const int* in_sizes, int n_in,
                              void* d_out, int out_size, void* d_ws, size_t ws_size,
                              hipStream_t stream) {
  const int*   ids   = (const int*)d_in[0];
  const int*   amask = (const int*)d_in[1];
  const float* wemb  = (const float*)d_in[2];
  const float* pemb  = (const float*)d_in[3];
  const float* temb  = (const float*)d_in[4];
  const float* elns  = (const float*)d_in[5];
  const float* elnb  = (const float*)d_in[6];
  const float* Wq = (const float*)d_in[7];   const float* bq  = (const float*)d_in[8];
  const float* Wk = (const float*)d_in[9];   const float* bk  = (const float*)d_in[10];
  const float* Wv = (const float*)d_in[11];  const float* bv  = (const float*)d_in[12];
  const float* Wao = (const float*)d_in[13]; const float* bao = (const float*)d_in[14];
  const float* l1s = (const float*)d_in[15]; const float* l1b = (const float*)d_in[16];
  const float* Wi = (const float*)d_in[17];  const float* bi  = (const float*)d_in[18];
  const float* Wmo = (const float*)d_in[19]; const float* bmo = (const float*)d_in[20];
  const float* l2s = (const float*)d_in[21]; const float* l2b = (const float*)d_in[22];

  char* ws = (char*)d_ws;
  size_t off = 0;
  auto alloc = [&](size_t bytes) -> void* {
    void* p = ws + off;
    off += (bytes + 255) & ~(size_t)255;
    return p;
  };

  const size_t nDD  = (size_t)LL * DD * DD;    // per-weight-set elements
  const size_t nDF  = (size_t)LL * DD * FFD;
  __bf16* wqb  = (__bf16*)alloc(nDD * 2);
  __bf16* wkb  = (__bf16*)alloc(nDD * 2);
  __bf16* wvb  = (__bf16*)alloc(nDD * 2);
  __bf16* waob = (__bf16*)alloc(nDD * 2);
  __bf16* wib  = (__bf16*)alloc(nDF * 2);
  __bf16* wmob = (__bf16*)alloc(nDF * 2);
  float*  hF   = (float*) alloc((size_t)MR * DD * 4);
  __bf16* hB   = (__bf16*)alloc((size_t)MR * DD * 2);
  float*  h2F  = (float*) alloc((size_t)MR * DD * 4);
  __bf16* h2B  = (__bf16*)alloc((size_t)MR * DD * 2);
  __bf16* qB   = (__bf16*)alloc((size_t)MR * DD * 2);
  __bf16* kB   = (__bf16*)alloc((size_t)MR * DD * 2);
  __bf16* vB   = (__bf16*)alloc((size_t)MR * DD * 2);
  __bf16* ctxB = (__bf16*)alloc((size_t)MR * DD * 2);
  float*  tmpF = (float*) alloc((size_t)MR * DD * 4);
  __bf16* ffnB = (__bf16*)alloc((size_t)MR * FFD * 2);
  (void)ws_size; (void)n_in; (void)in_sizes; (void)out_size;

  auto cvt = [&](const float* src, __bf16* dst, size_t n) {
    size_t blocks = (n + 255) / 256;
    if (blocks > 8192) blocks = 8192;
    f32_to_bf16_k<<<dim3((unsigned)blocks), 256, 0, stream>>>(src, dst, n);
  };
  cvt(Wq, wqb, nDD);  cvt(Wk, wkb, nDD);  cvt(Wv, wvb, nDD);
  cvt(Wao, waob, nDD); cvt(Wi, wib, nDF); cvt(Wmo, wmob, nDF);

  embed_ln_k<<<MR, 256, 0, stream>>>(ids, wemb, pemb, temb, elns, elnb, hF, hB);

  const dim3 gD (MR / 64, DD  / 64);   // N = 768 GEMMs
  const dim3 gFF(MR / 64, FFD / 64);   // N = 3072 GEMM
  const dim3 gAT(SS / 128, HH, BB);

  for (int l = 0; l < LL; ++l) {
    const __bf16* wq = wqb  + (size_t)l * DD * DD;
    const __bf16* wk = wkb  + (size_t)l * DD * DD;
    const __bf16* wv = wvb  + (size_t)l * DD * DD;
    const __bf16* wo = waob + (size_t)l * DD * DD;
    const __bf16* wi = wib  + (size_t)l * DD * FFD;
    const __bf16* wm = wmob + (size_t)l * FFD * DD;

    gemm_bf16_k<0, false, true><<<gD, 128, 0, stream>>>(
        hB, wq, bq + (size_t)l * DD, nullptr, qB, MR, DD, DD);
    gemm_bf16_k<0, false, true><<<gD, 128, 0, stream>>>(
        hB, wk, bk + (size_t)l * DD, nullptr, kB, MR, DD, DD);
    gemm_bf16_k<0, false, true><<<gD, 128, 0, stream>>>(
        hB, wv, bv + (size_t)l * DD, nullptr, vB, MR, DD, DD);

    attention_k<<<gAT, 256, 0, stream>>>(qB, kB, vB, amask, ctxB);

    gemm_bf16_k<0, true, false><<<gD, 128, 0, stream>>>(
        ctxB, wo, bao + (size_t)l * DD, tmpF, nullptr, MR, DD, DD);
    residual_ln_k<<<MR, 256, 0, stream>>>(tmpF, hF, l1s + (size_t)l * DD,
                                          l1b + (size_t)l * DD, h2F, h2B);

    gemm_bf16_k<1, false, true><<<gFF, 128, 0, stream>>>(
        h2B, wi, bi + (size_t)l * FFD, nullptr, ffnB, MR, FFD, DD);
    gemm_bf16_k<0, true, false><<<gD, 128, 0, stream>>>(
        ffnB, wm, bmo + (size_t)l * DD, tmpF, nullptr, MR, DD, FFD);
    residual_ln_k<<<MR, 256, 0, stream>>>(tmpF, h2F, l2s + (size_t)l * DD,
                                          l2b + (size_t)l * DD, hF, hB);
  }

  pool_norm_k<<<BB, 256, 0, stream>>>(hF, amask, (float*)d_out);
}
